// SAE_38963943309813
// MI455X (gfx1250) — matrix operand order repaired and checked by reading, not compiled, not run on hardware
//
#include <hip/hip_runtime.h>

// Problem constants (from reference)
#define BB   4
#define CC   512
#define CI_  256
#define NN   4096   // H*W = 64*64

typedef __attribute__((ext_vector_type(16))) __bf16 v16bf;
typedef __attribute__((ext_vector_type(8)))  float  v8f;

union AV { v16bf v; uint4 q[2]; };   // 32B WMMA bf16 operand (A or B)
union F8 { v8f   v; float f[8]; };   // 32B f32 C/D accumulator

__device__ __forceinline__ v8f wmma_bf16(v16bf a, v16bf b, v8f c) {
  // D = A(16x32) * B(32x16) + C(16x16 f32)
  return __builtin_amdgcn_wmma_f32_16x16x32_bf16(
      /*neg_a=*/false, a, /*neg_b=*/false, b,
      /*c_mod=*/(short)0, c, /*reuse_a=*/false, /*reuse_b=*/false);
}

// ---------------------------------------------------------------- utilities
__global__ __launch_bounds__(256) void cvt_kernel(const float* __restrict__ s,
                                                  __bf16* __restrict__ d, int n) {
  for (int i = blockIdx.x * 256 + threadIdx.x; i < n; i += gridDim.x * 256)
    d[i] = (__bf16)s[i];
}

// xb[b][n][c] = bf16(x[b][c][n])
__global__ __launch_bounds__(256) void transpose_kernel(const float* __restrict__ x,
                                                        __bf16* __restrict__ xb) {
  __shared__ float t[32][33];
  const int b  = blockIdx.z;
  const int c0 = blockIdx.y * 32, n0 = blockIdx.x * 32;
  const int tx = threadIdx.x, ty = threadIdx.y;  // 32 x 8
  #pragma unroll
  for (int i = 0; i < 32; i += 8)
    t[ty + i][tx] = x[((long)b * CC + c0 + ty + i) * NN + n0 + tx];
  __syncthreads();
  #pragma unroll
  for (int i = 0; i < 32; i += 8)
    xb[((long)b * NN + n0 + ty + i) * CC + c0 + tx] = (__bf16)t[tx][ty + i];
}

// ------------------------------------------------------- projection GEMM
// out[n,o] = sum_c A[n,c] * W[o,c] + bias[o]   (A = xb, pixel-major)
// MODE 0: store bf16 row-major [B*N][Cout], value scaled by `scale`  (q / k)
// MODE 1: store bf16 channel-major [B][Cout][NN]                     (v)
template <int MODE>
__global__ __launch_bounds__(256) void proj_kernel(
    const __bf16* __restrict__ A,    // [B*N][CC]
    const __bf16* __restrict__ Wb,   // [Cout][CC]
    const float*  __restrict__ bias, // [Cout]
    int Cout, float scale,
    __bf16* __restrict__ outRM,      // MODE 0
    __bf16* __restrict__ outT)       // MODE 1
{
  const int lane = threadIdx.x & 31, hi = lane >> 4, lo = lane & 15;
  const int cgc = Cout >> 6;                       // channel groups of 64
  const int wid = blockIdx.x * 8 + (threadIdx.x >> 5);
  const long pix = (long)(wid / cgc) * 16;
  const int  cg  = wid % cgc;

  F8 acc[4];
  #pragma unroll
  for (int t = 0; t < 4; ++t)
    #pragma unroll
    for (int j = 0; j < 8; ++j) acc[t].f[j] = 0.f;

  const __bf16* arow = A + (pix + lo) * CC;        // A lane = matrix row (pixel)
  for (int kc = 0; kc < CC / 32; ++kc) {
    AV a;  // A 16x32: lane row=lo, K = hi*16..+15 contiguous
    a.q[0] = *(const uint4*)(arow + kc * 32 + hi * 16);
    a.q[1] = *(const uint4*)(arow + kc * 32 + hi * 16 + 8);
    #pragma unroll
    for (int t = 0; t < 4; ++t) {
      // B 32x16: lane col=lo (out-channel), K = hi*16..+15 contiguous
      const __bf16* wr = Wb + (long)(cg * 64 + t * 16 + lo) * CC + kc * 32 + hi * 16;
      AV bm;
      bm.q[0] = *(const uint4*)(wr);
      bm.q[1] = *(const uint4*)(wr + 8);
      acc[t].v = wmma_bf16(a.v, bm.v, acc[t].v);
    }
  }

  #pragma unroll
  for (int t = 0; t < 4; ++t) {
    const int ch = cg * 64 + t * 16 + lo;          // D: n = lane&15 -> channel
    const float bv = bias[ch];
    if (MODE == 0) {
      #pragma unroll
      for (int j = 0; j < 8; ++j) {                // D: m = vgpr + 8*hi -> pixel
        long row = pix + j + 8 * hi;
        outRM[row * Cout + ch] = (__bf16)((acc[t].f[j] + bv) * scale);
      }
    } else {
      const long p0 = pix + 8 * hi;
      const int  b  = (int)(p0 / NN);
      const int  n0 = (int)(p0 % NN);
      union { __bf16 h[8]; uint4 u; } pk;
      #pragma unroll
      for (int j = 0; j < 8; ++j) pk.h[j] = (__bf16)(acc[t].f[j] + bv);
      *(uint4*)(outT + ((long)b * Cout + ch) * NN + n0) = pk.u;   // contiguous pixels
    }
  }
}

// ------------------------------------------------------- flash attention
// block = 256 thr = 8 waves: qt = wave>>2 (two 16-query tiles), cg = wave&3
// (four 128-channel groups). cg==0 waves compute scores + online softmax.
__global__ __launch_bounds__(256) void flash_kernel(
    const __bf16* __restrict__ qb,   // [B*N][CI] (pre-scaled by Ci^-0.5)
    const __bf16* __restrict__ kb,   // [B*N][CI]
    const __bf16* __restrict__ vb,   // [B][CC][NN]
    __bf16* __restrict__ ob)         // [B*N][CC]
{
  __shared__ __align__(16) __bf16 Pl[2][16 * 32];  // P tile, row-major [16 q][32 k]
  __shared__ __align__(16) float  Cl[2][16];       // per-row rescale factor
  __shared__ __align__(16) float  Ll[2][16];       // final per-row softmax sum

  const int lane = threadIdx.x & 31, hi = lane >> 4, lo = lane & 15;
  const int wv = threadIdx.x >> 5;
  const int qt = wv >> 2, cg = wv & 3;
  const int b  = blockIdx.x >> 7;                  // NN/32 = 128 blocks per batch
  const int q0 = (blockIdx.x & 127) * 32 + qt * 16;
  const long pixg = (long)b * NN + q0;

  AV aQ[8];                                        // Q as A-operand, Ci=256 = 8 K-chunks
  if (cg == 0) {
    const __bf16* qrow = qb + (pixg + lo) * CI_;
    #pragma unroll
    for (int kc = 0; kc < 8; ++kc) {
      aQ[kc].q[0] = *(const uint4*)(qrow + kc * 32 + hi * 16);
      aQ[kc].q[1] = *(const uint4*)(qrow + kc * 32 + hi * 16 + 8);
    }
  }

  F8 O[8];                                         // 16 queries x 128 channels, f32
  #pragma unroll
  for (int t = 0; t < 8; ++t)
    #pragma unroll
    for (int j = 0; j < 8; ++j) O[t].f[j] = 0.f;
  float mr[8], lr[8];
  #pragma unroll
  for (int j = 0; j < 8; ++j) { mr[j] = -1e30f; lr[j] = 0.f; }

  const int chb = cg * 128;
  const __bf16* vbb = vb + (long)b * CC * NN;

  for (int k0 = 0; k0 < NN; k0 += 32) {
    if (cg == 0) {
      F8 S0, S1;
      #pragma unroll
      for (int j = 0; j < 8; ++j) { S0.f[j] = 0.f; S1.f[j] = 0.f; }
      const __bf16* kr0 = kb + ((long)b * NN + k0 + lo) * CI_;   // B lane = key column
      const __bf16* kr1 = kr0 + 16 * CI_;
      #pragma unroll
      for (int kc = 0; kc < 8; ++kc) {
        AV b0, b1;
        b0.q[0] = *(const uint4*)(kr0 + kc * 32 + hi * 16);
        b0.q[1] = *(const uint4*)(kr0 + kc * 32 + hi * 16 + 8);
        b1.q[0] = *(const uint4*)(kr1 + kc * 32 + hi * 16);
        b1.q[1] = *(const uint4*)(kr1 + kc * 32 + hi * 16 + 8);
        S0.v = wmma_bf16(aQ[kc].v, b0.v, S0.v);
        S1.v = wmma_bf16(aQ[kc].v, b1.v, S1.v);
      }
      if (k0 + 32 < NN) __builtin_prefetch(kr0 + 32 * CI_, 0, 0);
      // online softmax; D layout: row m = r + 8*hi lives on the 16-lane half
      #pragma unroll
      for (int r = 0; r < 8; ++r) {
        float mx = fmaxf(S0.f[r], S1.f[r]);
        #pragma unroll
        for (int d = 1; d < 16; d <<= 1) mx = fmaxf(mx, __shfl_xor(mx, d, 32));
        const float mn = fmaxf(mr[r], mx);
        const float c  = __expf(mr[r] - mn);
        const float p0 = __expf(S0.f[r] - mn);
        const float p1 = __expf(S1.f[r] - mn);
        float s = p0 + p1;
        #pragma unroll
        for (int d = 1; d < 16; d <<= 1) s += __shfl_xor(s, d, 32);
        lr[r] = lr[r] * c + s;
        mr[r] = mn;
        Pl[qt][(r + 8 * hi) * 32 + lo]      = (__bf16)p0;
        Pl[qt][(r + 8 * hi) * 32 + 16 + lo] = (__bf16)p1;
        if (lo == 0) Cl[qt][8 * hi + r] = c;
      }
    }
    __syncthreads();

    // all 8 waves: rescale O, then P·V for 8 channel tiles
    float cf[8];
    {
      const float4* cp = (const float4*)&Cl[qt][8 * hi];
      float4 c0 = cp[0], c1 = cp[1];
      cf[0] = c0.x; cf[1] = c0.y; cf[2] = c0.z; cf[3] = c0.w;
      cf[4] = c1.x; cf[5] = c1.y; cf[6] = c1.z; cf[7] = c1.w;
    }
    #pragma unroll
    for (int t = 0; t < 8; ++t)
      #pragma unroll
      for (int j = 0; j < 8; ++j) O[t].f[j] *= cf[j];

    AV aP;                                        // P as A-operand from LDS
    aP.q[0] = *(const uint4*)&Pl[qt][lo * 32 + hi * 16];
    aP.q[1] = *(const uint4*)&Pl[qt][lo * 32 + hi * 16 + 8];
    #pragma unroll
    for (int t = 0; t < 8; ++t) {
      // V as B-operand: lane = channel column, 16 contiguous keys
      const __bf16* vr = vbb + (long)(chb + t * 16 + lo) * NN + k0 + hi * 16;
      AV bv_;
      bv_.q[0] = *(const uint4*)(vr);
      bv_.q[1] = *(const uint4*)(vr + 8);
      O[t].v = wmma_bf16(aP.v, bv_.v, O[t].v);
    }
    __syncthreads();
  }

  if (cg == 0 && lo == 0) {
    #pragma unroll
    for (int r = 0; r < 8; ++r) Ll[qt][8 * hi + r] = lr[r];
  }
  __syncthreads();
  float li[8];
  {
    const float4* lp = (const float4*)&Ll[qt][8 * hi];
    float4 l0 = lp[0], l1 = lp[1];
    li[0] = 1.f / l0.x; li[1] = 1.f / l0.y; li[2] = 1.f / l0.z; li[3] = 1.f / l0.w;
    li[4] = 1.f / l1.x; li[5] = 1.f / l1.y; li[6] = 1.f / l1.z; li[7] = 1.f / l1.w;
  }
  #pragma unroll
  for (int t = 0; t < 8; ++t) {
    const int ch = chb + t * 16 + lo;
    #pragma unroll
    for (int j = 0; j < 8; ++j)
      ob[(pixg + j + 8 * hi) * CC + ch] = (__bf16)(O[t].f[j] * li[j]);
  }
}

// ------------------------------------------- final Wo GEMM + gamma*out + x
__global__ __launch_bounds__(256) void final_kernel(
    const __bf16* __restrict__ A,    // ob [B*N][CC]
    const __bf16* __restrict__ Wb,   // Wo bf16 [CC][CC]
    const float*  __restrict__ bias, // bo
    const float*  __restrict__ gamma,
    const float*  __restrict__ x,    // [B][CC][NN]
    float*        __restrict__ y)    // [B][CC][NN]
{
  const int lane = threadIdx.x & 31, hi = lane >> 4, lo = lane & 15;
  const int wid = blockIdx.x * 8 + (threadIdx.x >> 5);
  const long pix = (long)(wid >> 3) * 16;          // 8 channel groups of 64
  const int  cg  = wid & 7;
  const float g = gamma[0];

  F8 acc[4];
  #pragma unroll
  for (int t = 0; t < 4; ++t)
    #pragma unroll
    for (int j = 0; j < 8; ++j) acc[t].f[j] = 0.f;

  const __bf16* arow = A + (pix + lo) * CC;
  for (int kc = 0; kc < CC / 32; ++kc) {
    AV a;
    a.q[0] = *(const uint4*)(arow + kc * 32 + hi * 16);
    a.q[1] = *(const uint4*)(arow + kc * 32 + hi * 16 + 8);
    #pragma unroll
    for (int t = 0; t < 4; ++t) {
      const __bf16* wr = Wb + (long)(cg * 64 + t * 16 + lo) * CC + kc * 32 + hi * 16;
      AV bm;
      bm.q[0] = *(const uint4*)(wr);
      bm.q[1] = *(const uint4*)(wr + 8);
      acc[t].v = wmma_bf16(a.v, bm.v, acc[t].v);
    }
  }

  const long p0 = pix + 8 * hi;
  const int  b  = (int)(p0 / NN);
  const int  n0 = (int)(p0 % NN);
  #pragma unroll
  for (int t = 0; t < 4; ++t) {
    const int ch = cg * 64 + t * 16 + lo;
    const float bv = bias[ch];
    const float* xr = x + ((long)b * CC + ch) * NN + n0;
    float4 x0 = *(const float4*)(xr);
    float4 x1 = *(const float4*)(xr + 4);
    float4 r0, r1;
    r0.x = g * (acc[t].f[0] + bv) + x0.x;
    r0.y = g * (acc[t].f[1] + bv) + x0.y;
    r0.z = g * (acc[t].f[2] + bv) + x0.z;
    r0.w = g * (acc[t].f[3] + bv) + x0.w;
    r1.x = g * (acc[t].f[4] + bv) + x1.x;
    r1.y = g * (acc[t].f[5] + bv) + x1.y;
    r1.z = g * (acc[t].f[6] + bv) + x1.z;
    r1.w = g * (acc[t].f[7] + bv) + x1.w;
    float* yr = y + ((long)b * CC + ch) * NN + n0;
    *(float4*)(yr)     = r0;
    *(float4*)(yr + 4) = r1;
  }
}

// ---------------------------------------------------------------- launcher
extern "C" void kernel_launch(void* const* d_in, const int* in_sizes, int n_in,
                              void* d_out, int out_size, void* d_ws, size_t ws_size,
                              hipStream_t stream) {
  (void)in_sizes; (void)n_in; (void)out_size; (void)ws_size;
  const float* x     = (const float*)d_in[0];
  const float* Wq    = (const float*)d_in[1];
  const float* bq    = (const float*)d_in[2];
  const float* Wk    = (const float*)d_in[3];
  const float* bk    = (const float*)d_in[4];
  const float* Wv    = (const float*)d_in[5];
  const float* bvp   = (const float*)d_in[6];
  const float* Wo    = (const float*)d_in[7];
  const float* bo    = (const float*)d_in[8];
  const float* gamma = (const float*)d_in[9];

  char* ws = (char*)d_ws;
  size_t off = 0;
  auto alloc = [&](size_t bytes) -> void* {
    void* p = ws + off;
    off += (bytes + 255) & ~(size_t)255;
    return p;
  };
  __bf16* xb  = (__bf16*)alloc((size_t)BB * NN * CC * 2);   // x transposed bf16
  __bf16* qb  = (__bf16*)alloc((size_t)BB * NN * CI_ * 2);  // q [B*N][Ci]
  __bf16* kbf = (__bf16*)alloc((size_t)BB * NN * CI_ * 2);  // k [B*N][Ci]
  __bf16* vbf = (__bf16*)alloc((size_t)BB * CC * NN * 2);   // v [B][C][N]
  __bf16* ob  = (__bf16*)alloc((size_t)BB * NN * CC * 2);   // attn out [B*N][C]
  __bf16* wqb = (__bf16*)alloc((size_t)CI_ * CC * 2);
  __bf16* wkb = (__bf16*)alloc((size_t)CI_ * CC * 2);
  __bf16* wvb = (__bf16*)alloc((size_t)CC * CC * 2);
  __bf16* wob = (__bf16*)alloc((size_t)CC * CC * 2);

  cvt_kernel<<<512, 256, 0, stream>>>(Wq, wqb, CI_ * CC);
  cvt_kernel<<<512, 256, 0, stream>>>(Wk, wkb, CI_ * CC);
  cvt_kernel<<<1024, 256, 0, stream>>>(Wv, wvb, CC * CC);
  cvt_kernel<<<1024, 256, 0, stream>>>(Wo, wob, CC * CC);

  transpose_kernel<<<dim3(NN / 32, CC / 32, BB), dim3(32, 8), 0, stream>>>(x, xb);

  // q/k: 4096 waves -> 512 blocks ; v: 8192 waves -> 1024 blocks
  const float qscale = 0.0625f;  // Ci^-0.5, folded into q
  proj_kernel<0><<<512, 256, 0, stream>>>(xb, wqb, bq, CI_, qscale, qb, nullptr);
  proj_kernel<0><<<512, 256, 0, stream>>>(xb, wkb, bk, CI_, 1.0f, kbf, nullptr);
  proj_kernel<1><<<1024, 256, 0, stream>>>(xb, wvb, bvp, CC, 1.0f, nullptr, vbf);

  flash_kernel<<<BB * (NN / 32), 256, 0, stream>>>(qb, kbf, vbf, ob);

  final_kernel<<<1024, 256, 0, stream>>>(ob, wob, bo, gamma, x, (float*)d_out);
}
